// DecoderModel_11141145165923
// MI455X (gfx1250) — matrix operand order repaired and checked by reading, not compiled, not run on hardware
//
#include <hip/hip_runtime.h>

#define LNUM 2
#define BDIM 2
#define SEQ 1024
#define HDIM 2048
#define NHEADS 16
#define HEADD 128
#define VOCAB 32000
#define PDIM 5632
#define MSQ SEQ
#define BSROWS (BDIM * SEQ)   /* 2048 token rows */
#define NEGV (-10000.0f)

typedef __bf16 bf16;
typedef __attribute__((ext_vector_type(16))) bf16  v16bf;
typedef __attribute__((ext_vector_type(8)))  float v8f;

__device__ __forceinline__ v8f wmma_bf16(v16bf a, v16bf b, v8f c) {
  return __builtin_amdgcn_wmma_f32_16x16x32_bf16(false, a, false, b, (short)0, c,
                                                 false, false);
}
__device__ __forceinline__ v8f vzero8() {
  v8f z;
#pragma unroll
  for (int i = 0; i < 8; ++i) z[i] = 0.f;
  return z;
}

// 32-bit LDS byte offset of a __shared__ object (generic -> AS3 -> int)
__device__ __forceinline__ unsigned lds_addr(const void* p) {
  return (unsigned)(size_t)(__attribute__((address_space(3))) const char*)p;
}
// CDNA5 async DMA: 16B global -> LDS per lane, tracked by ASYNCcnt
__device__ __forceinline__ void async_copy_b128(unsigned lds, const void* g) {
  asm volatile("global_load_async_to_lds_b128 %0, %1, off" ::"v"(lds), "v"(g)
               : "memory");
}
__device__ __forceinline__ void wait_async0() {
  asm volatile("s_wait_asynccnt 0x0" ::: "memory");
}

// ---------------------------------------------------------------------------
// Embedding gather: x[row, :] = tok_emb[tokens[row], :]
// ---------------------------------------------------------------------------
__global__ void embed_kernel(const int* __restrict__ tokens,
                             const float* __restrict__ emb,
                             float* __restrict__ x) {
  const int row = blockIdx.x;
  const int tid = threadIdx.x;
  const int tok = tokens[row];
  const float4* s = (const float4*)(emb + (size_t)tok * HDIM + tid * 8);
  float4* d = (float4*)(x + (size_t)row * HDIM + tid * 8);
  d[0] = s[0];
  d[1] = s[1];
}

// ---------------------------------------------------------------------------
// RMSNorm over H=2048: grid = rows, block = 256, 8 elements / thread
// ---------------------------------------------------------------------------
template <typename OutT>
__global__ void rmsnorm_kernel(const float* __restrict__ x,
                               const float* __restrict__ w,
                               OutT* __restrict__ out, float eps) {
  __shared__ float red[256];
  const int row = blockIdx.x;
  const int tid = threadIdx.x;
  const float* xr = x + (size_t)row * HDIM;
  float vals[8];
  float ss = 0.f;
#pragma unroll
  for (int j = 0; j < 8; ++j) {
    vals[j] = xr[tid * 8 + j];
    ss += vals[j] * vals[j];
  }
  red[tid] = ss;
  __syncthreads();
  for (int s = 128; s > 0; s >>= 1) {
    if (tid < s) red[tid] += red[tid + s];
    __syncthreads();
  }
  const float inv = rsqrtf(red[0] * (1.f / HDIM) + eps);
#pragma unroll
  for (int j = 0; j < 8; ++j)
    out[(size_t)row * HDIM + tid * 8 + j] = (OutT)(w[tid * 8 + j] * vals[j] * inv);
}

// Final RMSNorm on last-token rows only: grid = BDIM
__global__ void final_rms_kernel(const float* __restrict__ x,
                                 const float* __restrict__ w,
                                 float* __restrict__ out, float eps) {
  __shared__ float red[256];
  const int b = blockIdx.x;
  const int tid = threadIdx.x;
  const float* xr = x + ((size_t)b * SEQ + (SEQ - 1)) * HDIM;
  float vals[8];
  float ss = 0.f;
#pragma unroll
  for (int j = 0; j < 8; ++j) {
    vals[j] = xr[tid * 8 + j];
    ss += vals[j] * vals[j];
  }
  red[tid] = ss;
  __syncthreads();
  for (int s = 128; s > 0; s >>= 1) {
    if (tid < s) red[tid] += red[tid + s];
    __syncthreads();
  }
  const float inv = rsqrtf(red[0] * (1.f / HDIM) + eps);
#pragma unroll
  for (int j = 0; j < 8; ++j)
    out[(size_t)b * HDIM + tid * 8 + j] = w[tid * 8 + j] * vals[j] * inv;
}

// ---------------------------------------------------------------------------
// GEMM: C[M,N] = A[M,K](bf16) * W[N,K](fp32->bf16)^T  (+ residual Cin)
// 256 threads = 8 waves, tile 128x128x32; wave = 32x64 (2x4 WMMA frags).
// Double-buffered LDS; A tile via async-to-LDS DMA, W tile reg-staged + cvt.
// ---------------------------------------------------------------------------
#define SA_STRIDE 40  /* bf16 elems; 80B row stride -> 16B aligned, no conflicts */
template <bool RES>
__global__ __launch_bounds__(256) void gemm_bf16_kernel(
    const bf16* __restrict__ A, const float* __restrict__ W,
    const float* __restrict__ Cin, float* __restrict__ Cout, int M, int N,
    int K) {
  __shared__ bf16 sA[2][128 * SA_STRIDE];
  __shared__ bf16 sB[2][128 * SA_STRIDE];
  const int tid = threadIdx.x;
  const int wave = tid >> 5, lane = tid & 31;
  const int half = lane >> 4, l16 = lane & 15;
  const int wm = wave >> 1, wn = wave & 1;  // 4 x 2 wave grid
  const int m0 = blockIdx.y * 128, n0 = blockIdx.x * 128;

  v8f acc[2][4];
#pragma unroll
  for (int i = 0; i < 2; ++i)
#pragma unroll
    for (int j = 0; j < 4; ++j) acc[i][j] = vzero8();

  const int r = tid >> 1, seg = (tid & 1) * 16;
  const bf16* aRow = A + (size_t)(m0 + r) * K + seg;
  const float* wRow = W + (size_t)(n0 + r) * K + seg;
  const unsigned ldsA0 = lds_addr(&sA[0][r * SA_STRIDE + seg]);
  const unsigned ldsA1 = lds_addr(&sA[1][r * SA_STRIDE + seg]);

  float wreg[16];
  auto issueA = [&](int buf, int k0) {
    const unsigned dst = buf ? ldsA1 : ldsA0;
    async_copy_b128(dst, aRow + k0);
    async_copy_b128(dst + 16, aRow + k0 + 8);
  };
  auto loadW = [&](int k0) {
    const float4* w4 = (const float4*)(wRow + k0);
    float4 f0 = w4[0], f1 = w4[1], f2 = w4[2], f3 = w4[3];
    wreg[0] = f0.x;  wreg[1] = f0.y;  wreg[2] = f0.z;  wreg[3] = f0.w;
    wreg[4] = f1.x;  wreg[5] = f1.y;  wreg[6] = f1.z;  wreg[7] = f1.w;
    wreg[8] = f2.x;  wreg[9] = f2.y;  wreg[10] = f2.z; wreg[11] = f2.w;
    wreg[12] = f3.x; wreg[13] = f3.y; wreg[14] = f3.z; wreg[15] = f3.w;
  };
  auto storeB = [&](int buf) {
    bf16* bd = &sB[buf][r * SA_STRIDE + seg];
#pragma unroll
    for (int j = 0; j < 16; ++j) bd[j] = (bf16)wreg[j];
  };

  // prologue: stage tile 0
  issueA(0, 0);
  loadW(0);
  storeB(0);
  wait_async0();
  __syncthreads();

  const int nk = K / 32;
  for (int kt = 0; kt < nk; ++kt) {
    const int cur = kt & 1, nxt = cur ^ 1;
    const bool more = (kt + 1) < nk;
    if (more) {
      issueA(nxt, (kt + 1) * 32);  // async DMA into other buffer
      loadW((kt + 1) * 32);        // global loads in flight across compute
    }

    v16bf afr[2], bfr[4];
#pragma unroll
    for (int fi = 0; fi < 2; ++fi) {
      const int row = wm * 32 + fi * 16 + l16;
#pragma unroll
      for (int i = 0; i < 16; ++i) {
        const int kidx = i + half * 16;
        afr[fi][i] = sA[cur][row * SA_STRIDE + kidx];
      }
    }
#pragma unroll
    for (int fj = 0; fj < 4; ++fj) {
      const int col = wn * 64 + fj * 16 + l16;
#pragma unroll
      for (int i = 0; i < 16; ++i)
        bfr[fj][i] = sB[cur][col * SA_STRIDE + i + half * 16];
    }
#pragma unroll
    for (int fi = 0; fi < 2; ++fi)
#pragma unroll
      for (int fj = 0; fj < 4; ++fj)
        acc[fi][fj] = wmma_bf16(afr[fi], bfr[fj], acc[fi][fj]);

    if (more) storeB(nxt);
    wait_async0();
    __syncthreads();
  }

#pragma unroll
  for (int fi = 0; fi < 2; ++fi)
#pragma unroll
    for (int fj = 0; fj < 4; ++fj)
#pragma unroll
      for (int rr = 0; rr < 8; ++rr) {
        const int row = m0 + wm * 32 + fi * 16 + rr + half * 8;
        const int col = n0 + wn * 64 + fj * 16 + l16;
        const size_t idx = (size_t)row * N + col;
        float v = acc[fi][fj][rr];
        if (RES) v += Cin[idx];
        Cout[idx] = v;
      }
}

// ---------------------------------------------------------------------------
// RoPE + [B,S,NH,HD] -> [B,NH,S,HD] transpose; writes fp32 k/v caches to d_out
// and bf16 q/k/v for attention.  grid = B*NH*S, block = HEADD
// ---------------------------------------------------------------------------
__global__ void rope_kernel(const float* __restrict__ qf,
                            const float* __restrict__ kf,
                            const float* __restrict__ vf,
                            const int* __restrict__ pos_ids,
                            bf16* __restrict__ qbo, bf16* __restrict__ kbo,
                            bf16* __restrict__ vbo, float* __restrict__ kc_out,
                            float* __restrict__ vc_out) {
  const int idx = blockIdx.x;
  const int s = idx % SEQ;
  const int h = (idx / SEQ) % NHEADS;
  const int b = idx / (SEQ * NHEADS);
  const int i = threadIdx.x;
  const int pos = pos_ids[b * SEQ + s];
  const float fr = __powf(10000.f, -((float)(i & 63)) * (1.f / 64.f));
  const float ang = (float)pos * fr;
  const float c = __cosf(ang), sn = __sinf(ang);
  const size_t src = ((size_t)(b * SEQ + s) * NHEADS + h) * HEADD;
  const float qv = qf[src + i], kv = kf[src + i], vv = vf[src + i];
  const float qo = (i < 64) ? qv * c - qf[src + i + 64] * sn
                            : qv * c + qf[src + i - 64] * sn;
  const float ko = (i < 64) ? kv * c - kf[src + i + 64] * sn
                            : kv * c + kf[src + i - 64] * sn;
  const size_t dst = ((size_t)(b * NHEADS + h) * SEQ + s) * HEADD + i;
  qbo[dst] = (bf16)qo;
  kbo[dst] = (bf16)ko;
  vbo[dst] = (bf16)vv;
  kc_out[dst] = ko;
  vc_out[dst] = vv;
}

// ---------------------------------------------------------------------------
// Flash attention: grid (S/64, B*NH), block 128 (4 waves, 16 q-rows/wave).
// KV staged via async-to-LDS DMA in 32-wide chunks, online softmax fp32,
// QK^T and P*V via WMMA bf16.
// ---------------------------------------------------------------------------
#define SK_PAD 8  /* row stride (128+8)*2 = 272B -> 16B aligned */
__global__ __launch_bounds__(128) void flash_attn_kernel(
    const bf16* __restrict__ qb, const bf16* __restrict__ kb,
    const bf16* __restrict__ vb, const float* __restrict__ attn_mask,
    bf16* __restrict__ attnb) {
  const int bh = blockIdx.y;
  const int b = bh / NHEADS, h = bh % NHEADS;
  const int tid = threadIdx.x;
  const int wave = tid >> 5, lane = tid & 31;
  const int half = lane >> 4, l16 = lane & 15;
  const int qrow0 = blockIdx.x * 64 + wave * 16;

  __shared__ bf16 sk[32][HEADD + SK_PAD];
  __shared__ bf16 sv[32][HEADD + SK_PAD];
  __shared__ bf16 sp[4][16][32];

  const bf16* qptr = qb + ((size_t)(b * NHEADS + h)) * SEQ * HEADD;
  const bf16* kptr = kb + ((size_t)(b * NHEADS + h)) * SEQ * HEADD;
  const bf16* vptr = vb + ((size_t)(b * NHEADS + h)) * SEQ * HEADD;

  // Q fragments for this wave's 16 rows, K-dim = HD (4 frags of 32)
  v16bf aq[4];
  {
    const int row = qrow0 + l16;
#pragma unroll
    for (int kk = 0; kk < 4; ++kk)
#pragma unroll
      for (int i = 0; i < 16; ++i) {
        const int kd = kk * 32 + half * 16 + i;
        aq[kk][i] = qptr[(size_t)row * HEADD + kd];
      }
  }

  v8f o[8];
#pragma unroll
  for (int j = 0; j < 8; ++j) o[j] = vzero8();
  float mst[8], lst[8];
#pragma unroll
  for (int r = 0; r < 8; ++r) { mst[r] = -1e30f; lst[r] = 0.f; }

  const float sc = 0.08838834764831845f;  // 1/sqrt(128)
  const int kvend = (blockIdx.x + 1) * 64;  // causal upper bound

  // per-thread staging geometry: 32 rows x HEADD, 64B per thread per matrix
  const int sr = tid >> 2, sseg = (tid & 3) * 32;
  const unsigned dk0 = lds_addr(&sk[sr][sseg]);
  const unsigned dv0 = lds_addr(&sv[sr][sseg]);

  for (int kv0 = 0; kv0 < kvend; kv0 += 32) {
    {  // async stage K,V chunk into LDS
      const bf16* gk = kptr + (size_t)(kv0 + sr) * HEADD + sseg;
      const bf16* gv = vptr + (size_t)(kv0 + sr) * HEADD + sseg;
#pragma unroll
      for (int c = 0; c < 4; ++c) {
        async_copy_b128(dk0 + c * 16, gk + c * 8);
        async_copy_b128(dv0 + c * 16, gv + c * 8);
      }
      wait_async0();
    }
    __syncthreads();

    // S = Q * K^T : two 16x16 tiles covering 32 kv cols
    v8f s0 = vzero8(), s1 = vzero8();
#pragma unroll
    for (int kk = 0; kk < 4; ++kk) {
      v16bf b0, b1;
#pragma unroll
      for (int i = 0; i < 16; ++i) {
        const int kd = kk * 32 + half * 16 + i;
        b0[i] = sk[l16][kd];
        b1[i] = sk[16 + l16][kd];
      }
      s0 = wmma_bf16(aq[kk], b0, s0);
      s1 = wmma_bf16(aq[kk], b1, s1);
    }

    // scale + causal + pad mask
    const int c0 = kv0 + l16, c1 = kv0 + 16 + l16;
    const float pm0 = (attn_mask[b * MSQ + c0] == 0.f) ? NEGV : 0.f;
    const float pm1 = (attn_mask[b * MSQ + c1] == 0.f) ? NEGV : 0.f;
    float e0[8], e1[8], rmax[8];
#pragma unroll
    for (int r = 0; r < 8; ++r) {
      const int qr = qrow0 + r + half * 8;
      e0[r] = s0[r] * sc + ((c0 <= qr) ? 0.f : NEGV) + pm0;
      e1[r] = s1[r] * sc + ((c1 <= qr) ? 0.f : NEGV) + pm1;
      rmax[r] = fmaxf(e0[r], e1[r]);
    }
#pragma unroll
    for (int m = 1; m < 16; m <<= 1)
#pragma unroll
      for (int r = 0; r < 8; ++r)
        rmax[r] = fmaxf(rmax[r], __shfl_xor(rmax[r], m, 32));

    float rsum[8], fac[8];
#pragma unroll
    for (int r = 0; r < 8; ++r) {
      const float newm = fmaxf(mst[r], rmax[r]);
      fac[r] = __expf(mst[r] - newm);
      mst[r] = newm;
      e0[r] = __expf(e0[r] - newm);
      e1[r] = __expf(e1[r] - newm);
      rsum[r] = e0[r] + e1[r];
      sp[wave][r + half * 8][l16] = (bf16)e0[r];
      sp[wave][r + half * 8][16 + l16] = (bf16)e1[r];
    }
#pragma unroll
    for (int m = 1; m < 16; m <<= 1)
#pragma unroll
      for (int r = 0; r < 8; ++r) rsum[r] += __shfl_xor(rsum[r], m, 32);
#pragma unroll
    for (int r = 0; r < 8; ++r) lst[r] = lst[r] * fac[r] + rsum[r];
#pragma unroll
    for (int j = 0; j < 8; ++j)
#pragma unroll
      for (int r = 0; r < 8; ++r) o[j][r] *= fac[r];

    // wave-private P relayout through LDS: wait for our own ds stores
    asm volatile("s_wait_dscnt 0x0" ::: "memory");

    // O += P(16x32) * V(32x128)
    v16bf pa;
#pragma unroll
    for (int i = 0; i < 16; ++i)
      pa[i] = sp[wave][l16][i + half * 16];
#pragma unroll
    for (int j = 0; j < 8; ++j) {
      v16bf bv;
#pragma unroll
      for (int i = 0; i < 16; ++i) bv[i] = sv[i + half * 16][j * 16 + l16];
      o[j] = wmma_bf16(pa, bv, o[j]);
    }
    __syncthreads();
  }

  // normalize + write [B,S,H] bf16 (ready as GEMM A for output projection)
  float invl[8];
#pragma unroll
  for (int r = 0; r < 8; ++r) invl[r] = (lst[r] > 0.f) ? 1.f / lst[r] : 0.f;
#pragma unroll
  for (int j = 0; j < 8; ++j)
#pragma unroll
    for (int r = 0; r < 8; ++r) {
      const int row = qrow0 + r + half * 8;
      const int hd = j * 16 + l16;
      attnb[((size_t)b * SEQ + row) * HDIM + h * HEADD + hd] =
          (bf16)(o[j][r] * invl[r]);
    }
}

// ---------------------------------------------------------------------------
// SwiGLU: out = bf16( g*sigmoid(g)*u )
// ---------------------------------------------------------------------------
__global__ void swiglu_kernel(const float* __restrict__ g,
                              const float* __restrict__ u,
                              bf16* __restrict__ out) {
  const int i = blockIdx.x * blockDim.x + threadIdx.x;
  const float gv = g[i];
  const float s = 1.f / (1.f + __expf(-gv));
  out[i] = (bf16)(gv * s * u[i]);
}

// ---------------------------------------------------------------------------
// Logits GEMV (memory-bound on out_w, keep fp32): one wave per vocab row.
// ---------------------------------------------------------------------------
__global__ void logits_kernel(const float* __restrict__ xfin,
                              const float* __restrict__ out_w,
                              float* __restrict__ logits) {
  const int wave = threadIdx.x >> 5, lane = threadIdx.x & 31;
  const int rv = blockIdx.x * 8 + wave;
  const float* wr = out_w + (size_t)rv * HDIM;
  float a0 = 0.f, a1 = 0.f;
  for (int k = lane; k < HDIM; k += 32) {
    const float wv = wr[k];
    a0 += wv * xfin[k];
    a1 += wv * xfin[HDIM + k];
  }
#pragma unroll
  for (int m = 16; m > 0; m >>= 1) {
    a0 += __shfl_xor(a0, m, 32);
    a1 += __shfl_xor(a1, m, 32);
  }
  if (lane == 0) {
    logits[rv] = a0;
    logits[VOCAB + rv] = a1;
  }
}

__global__ void mask_kernel(const float* __restrict__ attn_mask,
                            float* __restrict__ out) {
  const int i = blockIdx.x * blockDim.x + threadIdx.x;
  const int b = i / MSQ, j = i % MSQ;
  out[i] = (j == MSQ - 1) ? 1.f : attn_mask[b * MSQ + j + 1];
}

// ---------------------------------------------------------------------------
extern "C" void kernel_launch(void* const* d_in, const int* in_sizes, int n_in,
                              void* d_out, int out_size, void* d_ws,
                              size_t ws_size, hipStream_t stream) {
  (void)in_sizes; (void)n_in; (void)out_size; (void)ws_size;

  const int* tokens = (const int*)d_in[0];
  const int* pos_ids = (const int*)d_in[1];
  const float* attn_mask = (const float*)d_in[2];
  // d_in[3] k_cache, d_in[4] v_cache unused (seq == MS => cache slice empty)
  const float* tok_emb = (const float*)d_in[5];
  const float* attn_nw = (const float*)d_in[6];
  const float* ffn_nw = (const float*)d_in[7];
  const float* wq = (const float*)d_in[8];
  const float* wk = (const float*)d_in[9];
  const float* wv = (const float*)d_in[10];
  const float* wo = (const float*)d_in[11];
  const float* w1 = (const float*)d_in[12];
  const float* w2 = (const float*)d_in[13];
  const float* w3 = (const float*)d_in[14];
  const float* fnw = (const float*)d_in[15];
  const float* out_w = (const float*)d_in[16];

  float* logits = (float*)d_out;                         // [B, V]
  float* mask_out = logits + (size_t)BDIM * VOCAB;       // [B, MS]
  float* kcs = mask_out + (size_t)BDIM * MSQ;            // [L,B,NH,MS,HD]
  float* vcs = kcs + (size_t)LNUM * BDIM * NHEADS * MSQ * HEADD;
  const size_t kv_layer = (size_t)BDIM * NHEADS * MSQ * HEADD;

  // workspace bump allocator (256B aligned)
  char* p = (char*)d_ws;
  auto alloc = [&](size_t bytes) -> char* {
    char* r = p;
    p += (bytes + 255) & ~(size_t)255;
    return r;
  };
  const size_t BH = (size_t)BSROWS * HDIM;
  float* x = (float*)alloc(BH * 4);
  float* qf = (float*)alloc(BH * 4);
  float* kf = (float*)alloc(BH * 4);
  float* vf = (float*)alloc(BH * 4);
  bf16* xn = (bf16*)alloc(BH * 2);
  bf16* qb = (bf16*)alloc(BH * 2);
  bf16* kb = (bf16*)alloc(BH * 2);
  bf16* vb = (bf16*)alloc(BH * 2);
  bf16* attnb = (bf16*)alloc(BH * 2);
  bf16* hn = (bf16*)alloc(BH * 2);
  float* gbuf = (float*)alloc((size_t)BSROWS * PDIM * 4);
  float* ubuf = (float*)alloc((size_t)BSROWS * PDIM * 4);
  bf16* gb = (bf16*)alloc((size_t)BSROWS * PDIM * 2);
  float* xfin = (float*)alloc((size_t)BDIM * HDIM * 4);

  embed_kernel<<<BSROWS, 256, 0, stream>>>(tokens, tok_emb, x);

  const dim3 gH(HDIM / 128, BSROWS / 128);
  const dim3 gP(PDIM / 128, BSROWS / 128);
  const dim3 gFA(SEQ / 64, BDIM * NHEADS);

  for (int l = 0; l < LNUM; ++l) {
    const size_t wofs = (size_t)l * HDIM * HDIM;
    const size_t pofs = (size_t)l * PDIM * HDIM;

    rmsnorm_kernel<bf16><<<BSROWS, 256, 0, stream>>>(x, attn_nw + l * HDIM, xn,
                                                     1e-6f);
    gemm_bf16_kernel<false><<<gH, 256, 0, stream>>>(xn, wq + wofs, nullptr, qf,
                                                    BSROWS, HDIM, HDIM);
    gemm_bf16_kernel<false><<<gH, 256, 0, stream>>>(xn, wk + wofs, nullptr, kf,
                                                    BSROWS, HDIM, HDIM);
    gemm_bf16_kernel<false><<<gH, 256, 0, stream>>>(xn, wv + wofs, nullptr, vf,
                                                    BSROWS, HDIM, HDIM);
    rope_kernel<<<BDIM * NHEADS * SEQ, HEADD, 0, stream>>>(
        qf, kf, vf, pos_ids, qb, kb, vb, kcs + l * kv_layer,
        vcs + l * kv_layer);
    flash_attn_kernel<<<gFA, 128, 0, stream>>>(qb, kb, vb, attn_mask, attnb);
    gemm_bf16_kernel<true><<<gH, 256, 0, stream>>>(attnb, wo + wofs, x, x,
                                                   BSROWS, HDIM, HDIM);
    rmsnorm_kernel<bf16><<<BSROWS, 256, 0, stream>>>(x, ffn_nw + l * HDIM, hn,
                                                     1e-6f);
    gemm_bf16_kernel<false><<<gP, 256, 0, stream>>>(hn, w1 + pofs, nullptr,
                                                    gbuf, BSROWS, PDIM, HDIM);
    gemm_bf16_kernel<false><<<gP, 256, 0, stream>>>(hn, w3 + pofs, nullptr,
                                                    ubuf, BSROWS, PDIM, HDIM);
    swiglu_kernel<<<(BSROWS * PDIM) / 256, 256, 0, stream>>>(gbuf, ubuf, gb);
    gemm_bf16_kernel<true><<<gH, 256, 0, stream>>>(gb, w2 + pofs, x, x, BSROWS,
                                                   HDIM, PDIM);
  }

  final_rms_kernel<<<BDIM, 256, 0, stream>>>(x, fnw, xfin, 1e-5f);
  logits_kernel<<<VOCAB / 8, 256, 0, stream>>>(xfin, out_w, logits);
  mask_kernel<<<(BDIM * MSQ) / 256, 256, 0, stream>>>(attn_mask, mask_out);
}
